// MultiHeadAttention_62251255988407
// MI455X (gfx1250) — matrix operand order repaired and checked by reading, not compiled, not run on hardware
//
#include <hip/hip_runtime.h>

// ---------------------------------------------------------------------------
// MultiHeadAttention forward for MI455X (gfx1250), bf16 WMMA pipeline.
//   k1: QKV projection   (x fp32 -> Q,K bf16 [B,H,S,64]; V transposed [B,H,64,S])
//   k2: flash attention  (TDM double-buffered K/V tiles, online softmax,
//                         v_wmma_f32_16x16x32_bf16 for QK^T and PV)
//   k3: output projection (Z bf16 x Wo -> fp32 out + bo; TDM A-tile load)
// Workspace: 4 * 16 MB bf16 tensors (Q, K, Vt, Z) = 64 MB.
// ---------------------------------------------------------------------------

#define B_SZ   4
#define S_LEN  2048
#define DM     1024
#define HEADS  16
#define DEPTH  64

typedef __attribute__((ext_vector_type(16))) __bf16 v16bf;
typedef __attribute__((ext_vector_type(8)))  __bf16 v8bf;
typedef __attribute__((ext_vector_type(8)))  float  v8f;
typedef __attribute__((ext_vector_type(4)))  unsigned int u32x4;
typedef __attribute__((ext_vector_type(8)))  int          i32x8;
typedef __attribute__((ext_vector_type(4)))  int          i32x4;

// Use the Tensor Data Mover when the toolchain exposes it (device pass only;
// host pass sees USE_TDM=0 and type-checks the fallback).
#if defined(__HIP_DEVICE_COMPILE__) && __has_builtin(__builtin_amdgcn_tensor_load_to_lds)
#define USE_TDM 1
#else
#define USE_TDM 0
#endif

__device__ __forceinline__ __bf16 f2bf(float f) {
    return (__bf16)f;   // native v_cvt (RNE) on gfx1250
}

__device__ __forceinline__ v8f wmma_bf16(v16bf a, v16bf b, v8f c) {
    // D = A(16x32 bf16) * B(32x16 bf16) + C(16x16 f32)
    return __builtin_amdgcn_wmma_f32_16x16x32_bf16(
        /*neg_a=*/false, a, /*neg_b=*/false, b,
        /*c_mod=*/(short)0, c, /*reuse_a=*/false, /*reuse_b=*/false);
}

// A-fragment (16x32 bf16) load from a row of >=32 contiguous bf16.
// Lane L holds M = L%16; element e -> K = e + (L>=16?16:0)
// => two contiguous 8-element (16B) chunks at K = base and K = base+8, base = (L>=16)?16:0.
__device__ __forceinline__ v16bf load_a_frag(const __bf16* row, int laneHi) {
    const int base = laneHi ? 16 : 0;
    v8bf lo = *(const v8bf*)(row + base);
    v8bf hi = *(const v8bf*)(row + base + 8);
    v16bf a;
#pragma unroll
    for (int i = 0; i < 8; ++i) { a[i] = lo[i]; a[i + 8] = hi[i]; }
    return a;
}

#if USE_TDM
// Issue a TDM 2D tile load (bf16 elements) into LDS. Builds the D# per
// CDNA5 ISA: group0 = {flags, lds_addr, global_addr, type},
// group1 = {mask/data_size, dims, tile dims, row stride}. Tracked by TENSORcnt.
__device__ __forceinline__ void tdm_load_2d_bf16(
    const void* gaddr, unsigned lds_byte_off,
    unsigned tile_d0, unsigned tile_d1,          // tile extents (elems; d0 contiguous)
    unsigned tensor_d0, unsigned tensor_d1,      // tensor extents (elems)
    unsigned long long row_stride)               // elems between rows (dim0 stride)
{
    unsigned long long ga = (unsigned long long)(size_t)gaddr;
    u32x4 g0;
    g0[0] = 1u;                                          // count=1 (valid, user mode)
    g0[1] = lds_byte_off;                                // lds_addr (bytes)
    g0[2] = (unsigned)(ga & 0xFFFFFFFFu);                // global_addr[31:0]
    g0[3] = (unsigned)((ga >> 32) & 0x01FFFFFFu) | (2u << 30);  // ga[56:32] | type=2
    i32x8 g1;
    g1[0] = (int)(1u << 16);                             // data_size=1 (2B), wg_mask=0
    g1[1] = (int)((tensor_d0 & 0xFFFFu) << 16);          // tensor_dim0[15:0] @ bits 63:48
    g1[2] = (int)(((tensor_d0 >> 16) & 0xFFFFu) |        // tensor_dim0[31:16]
                  ((tensor_d1 & 0xFFFFu) << 16));        // tensor_dim1[15:0]
    g1[3] = (int)(((tensor_d1 >> 16) & 0xFFFFu) |        // tensor_dim1[31:16]
                  ((tile_d0 & 0xFFFFu) << 16));          // tile_dim0
    g1[4] = (int)(tile_d1 & 0xFFFFu);                    // tile_dim1 (tile_dim2 = 0)
    g1[5] = (int)(unsigned)(row_stride & 0xFFFFFFFFull); // dim0_stride[31:0]
    g1[6] = (int)(unsigned)((row_stride >> 32) & 0xFFFFull); // dim0_stride[47:32]
    g1[7] = 0;
    i32x4 gz = {0, 0, 0, 0};
#if __clang_major__ >= 23
    i32x8 gz8 = {0, 0, 0, 0, 0, 0, 0, 0};
    __builtin_amdgcn_tensor_load_to_lds(g0, g1, gz, gz, gz8, 0);
#else
    __builtin_amdgcn_tensor_load_to_lds(g0, g1, gz, gz, 0);
#endif
}
__device__ __forceinline__ unsigned lds_off(const void* p) {
    return (unsigned)(size_t)p;   // flat shared ptr: LDS offset lives in low 32 bits
}
#endif

// ---------------------------------------------------------------------------
// Kernel 1: QKV projection.
// grid = (B*S/64, H, 3), block = 128 (4 waves). Each block: 64 rows x 64 cols.
// wave w computes col block [w*16, w*16+16) for all 64 rows (4 M-tiles).
// ---------------------------------------------------------------------------
__global__ __launch_bounds__(128) void qkv_proj_kernel(
    const float* __restrict__ x,
    const float* __restrict__ Wq, const float* __restrict__ bq,
    const float* __restrict__ Wk, const float* __restrict__ bk,
    const float* __restrict__ Wv, const float* __restrict__ bv,
    __bf16* __restrict__ Qb, __bf16* __restrict__ Kb, __bf16* __restrict__ Vt)
{
    __shared__ __attribute__((aligned(32))) __bf16 Xs[64][32];   // [row][k]
    __shared__ __attribute__((aligned(32))) __bf16 Wst[64][32];  // [col][k] (transposed)

    const int tid  = threadIdx.x;
    const int wave = tid >> 5;
    const int lane = tid & 31;
    const int l16  = lane & 15;
    const int hi   = lane >> 4;

    const int rtile = blockIdx.x;        // 64-row tile over B*S
    const int h     = blockIdx.y;
    const int which = blockIdx.z;        // 0=Q 1=K 2=V

    const float* W    = (which == 0) ? Wq : (which == 1) ? Wk : Wv;
    const float* bias = (which == 0) ? bq : (which == 1) ? bk : bv;

    v8f acc[4];
#pragma unroll
    for (int m = 0; m < 4; ++m) acc[m] = (v8f){0,0,0,0,0,0,0,0};

    for (int k0 = 0; k0 < DM; k0 += 32) {
        __syncthreads();
        // Xs: 64 rows x 32 fp32 -> bf16 (512 float4 chunks, 4 per thread)
#pragma unroll
        for (int i = 0; i < 4; ++i) {
            int t = tid + i * 128;                 // 0..511
            int r = t >> 3, seg = t & 7;
            float4 v = ((const float4*)(x + (size_t)(rtile * 64 + r) * DM + k0))[seg];
            __bf16* dst = &Xs[r][seg * 4];
            dst[0] = f2bf(v.x); dst[1] = f2bf(v.y);
            dst[2] = f2bf(v.z); dst[3] = f2bf(v.w);
        }
        // Wst[e][dl] = W[h][k0+dl][e]  (2048 elems, 16 per thread)
#pragma unroll
        for (int i = 0; i < 16; ++i) {
            int t = tid + i * 128;                 // 0..2047
            int e = t >> 5, dl = t & 31;
            Wst[e][dl] = f2bf(W[((size_t)h * DM + (k0 + dl)) * DEPTH + e]);
        }
        __syncthreads();

        // B frag: N = l16 (col), K = e + hi*16 -> contiguous 16 bf16 in Wst row
        v16bf bW = *(const v16bf*)&Wst[wave * 16 + l16][hi * 16];
#pragma unroll
        for (int m = 0; m < 4; ++m) {
            v16bf aX = load_a_frag(&Xs[m * 16 + l16][0], hi);
            acc[m] = wmma_bf16(aX, bW, acc[m]);
        }
    }

    const int col = wave * 16 + l16;                 // col within head [0,64)
    const float bv_ = bias[h * DEPTH + col];
#pragma unroll
    for (int m = 0; m < 4; ++m) {
#pragma unroll
        for (int r = 0; r < 8; ++r) {
            int R = rtile * 64 + m * 16 + r + hi * 8;  // global row in [0, B*S)
            int b = R >> 11, s = R & (S_LEN - 1);
            __bf16 ob = f2bf(acc[m][r] + bv_);
            size_t bh = (size_t)b * HEADS + h;
            if (which == 2)
                Vt[(bh * DEPTH + col) * S_LEN + s] = ob;     // [B,H,64,S]
            else if (which == 0)
                Qb[(bh * S_LEN + s) * DEPTH + col] = ob;     // [B,H,S,64]
            else
                Kb[(bh * S_LEN + s) * DEPTH + col] = ob;
        }
    }
}

// ---------------------------------------------------------------------------
// Kernel 2: flash attention.
// grid = (B*H, S/64), block = 128 (4 waves). Wave owns 16 query rows.
// Double-buffered LDS tiles; K/V^T tiles come in via the Tensor Data Mover.
// ---------------------------------------------------------------------------
__global__ __launch_bounds__(128) void attn_kernel(
    const __bf16* __restrict__ Qb, const __bf16* __restrict__ Kb,
    const __bf16* __restrict__ Vt, const float* __restrict__ mask,
    __bf16* __restrict__ Zb)
{
    __shared__ __attribute__((aligned(32))) __bf16 Ks[2][64][64];   // [buf][key][depth]
    __shared__ __attribute__((aligned(32))) __bf16 Vs[2][64][64];   // [buf][depth][key]
    __shared__ __attribute__((aligned(32))) __bf16 Ps[4][16][64];   // per-wave P staging
    __shared__ float mAdd[2][64];

    const int tid  = threadIdx.x;
    const int wave = tid >> 5;
    const int lane = tid & 31;
    const int l16  = lane & 15;
    const int hi   = lane >> 4;

    const int bh    = blockIdx.x;          // b*H + h
    const int b     = bh >> 4;
    const int h     = bh & 15;
    const int qtile = blockIdx.y;          // 64-query tile

    // Q A-fragments (held in registers for the whole K loop)
    const int qrow = qtile * 64 + wave * 16 + l16;
    const __bf16* qptr = Qb + ((size_t)bh * S_LEN + qrow) * DEPTH;
    v16bf aQ0 = load_a_frag(qptr + 0,  hi);
    v16bf aQ1 = load_a_frag(qptr + 32, hi);

    v8f O[4];
#pragma unroll
    for (int d = 0; d < 4; ++d) O[d] = (v8f){0,0,0,0,0,0,0,0};
    float mrow[8], lrow[8];
#pragma unroll
    for (int r = 0; r < 8; ++r) { mrow[r] = -3.0e38f; lrow[r] = 0.0f; }

    // Issue loads of key-block j into LDS buffer `buf`.
    auto issue_tiles = [&](int j, int buf) {
#if USE_TDM
        if (wave == 0) {
            // K tile: 64 keys x 64 depth, contiguous rows of Kb
            tdm_load_2d_bf16(Kb + ((size_t)bh * S_LEN + j * 64) * DEPTH,
                             lds_off(&Ks[buf][0][0]),
                             /*tile*/ 64, 64, /*tensor*/ 64, 64, /*stride*/ DEPTH);
            // V^T tile: 64 depth rows x 64 keys, rows strided by S in Vt
            tdm_load_2d_bf16(Vt + (size_t)bh * DEPTH * S_LEN + j * 64,
                             lds_off(&Vs[buf][0][0]),
                             /*tile*/ 64, 64, /*tensor*/ 64, 64, /*stride*/ S_LEN);
        }
#else
        {   // K tile: contiguous 8KB of Kb
            const uint4* src = (const uint4*)(Kb + ((size_t)bh * S_LEN + j * 64) * DEPTH);
            uint4* dst = (uint4*)&Ks[buf][0][0];
#pragma unroll
            for (int i = 0; i < 4; ++i) dst[tid + i * 128] = src[tid + i * 128];
        }
#pragma unroll
        for (int i = 0; i < 4; ++i) {   // V^T tile: 64 strided 128B rows
            int t = tid + i * 128;
            int row = t >> 3, seg = t & 7;
            ((uint4*)&Vs[buf][0][0])[t] =
                ((const uint4*)(Vt + ((size_t)bh * DEPTH + row) * S_LEN + j * 64))[seg];
        }
#endif
        if (tid < 64) {
            float mv = mask[(size_t)b * S_LEN + j * 64 + tid];
            mAdd[buf][tid] = (1.0f - mv) * -1.0e9f;
        }
    };

    issue_tiles(0, 0);

    for (int j = 0; j < S_LEN / 64; ++j) {
        const int buf = j & 1;
#if USE_TDM
        if (wave == 0) __builtin_amdgcn_s_wait_tensorcnt(0);
#endif
        __syncthreads();   // tiles for block j ready; prev compute done with buf^1
        if (j + 1 < S_LEN / 64) issue_tiles(j + 1, buf ^ 1);

        // ----- scores: S[16 x 64] = Q K^T * scale + mask -----
        v8f sc[4];
#pragma unroll
        for (int n = 0; n < 4; ++n) {
            int key = n * 16 + l16;
            v16bf b0 = *(const v16bf*)&Ks[buf][key][hi ? 16 : 0];
            v16bf b1 = *(const v16bf*)&Ks[buf][key][32 + (hi ? 16 : 0)];
            v8f c = (v8f){0,0,0,0,0,0,0,0};
            c = wmma_bf16(aQ0, b0, c);
            c = wmma_bf16(aQ1, b1, c);
            float add = mAdd[buf][key];
#pragma unroll
            for (int r = 0; r < 8; ++r) c[r] = c[r] * 0.125f + add;  // 1/sqrt(64)
            sc[n] = c;
        }

        // ----- online softmax (row = r + hi*8; reduce across 16-lane half) -----
#pragma unroll
        for (int r = 0; r < 8; ++r) {
            float rm = fmaxf(fmaxf(sc[0][r], sc[1][r]), fmaxf(sc[2][r], sc[3][r]));
#pragma unroll
            for (int off = 8; off > 0; off >>= 1)
                rm = fmaxf(rm, __shfl_xor(rm, off, 16));
            float mnew  = fmaxf(mrow[r], rm);
            float alpha = __expf(mrow[r] - mnew);
            float ps = 0.0f;
#pragma unroll
            for (int n = 0; n < 4; ++n) {
                float p = __expf(sc[n][r] - mnew);
                sc[n][r] = p;
                ps += p;
            }
#pragma unroll
            for (int off = 8; off > 0; off >>= 1)
                ps += __shfl_xor(ps, off, 16);
            lrow[r] = lrow[r] * alpha + ps;
            mrow[r] = mnew;
#pragma unroll
            for (int d = 0; d < 4; ++d) O[d][r] *= alpha;
        }

        // ----- stage P (C-layout -> row-major bf16 in per-wave LDS) -----
#pragma unroll
        for (int n = 0; n < 4; ++n) {
            int key = n * 16 + l16;
#pragma unroll
            for (int r = 0; r < 8; ++r)
                Ps[wave][r + hi * 8][key] = f2bf(sc[n][r]);
        }

        // ----- O += P V  (A from Ps, B from Vs; both contiguous per lane) -----
        const __bf16* prow = &Ps[wave][l16][0];
#pragma unroll
        for (int kc = 0; kc < 2; ++kc) {
            v16bf aP = load_a_frag(prow + kc * 32, hi);
#pragma unroll
            for (int dn = 0; dn < 4; ++dn) {
                v16bf bV = *(const v16bf*)&Vs[buf][dn * 16 + l16][kc * 32 + (hi ? 16 : 0)];
                O[dn] = wmma_bf16(aP, bV, O[dn]);
            }
        }
    }

    // ----- epilogue: O /= l, write Z in [B,S,H*64] bf16 -----
#pragma unroll
    for (int dn = 0; dn < 4; ++dn) {
#pragma unroll
        for (int r = 0; r < 8; ++r) {
            int M = r + hi * 8;
            int q = qtile * 64 + wave * 16 + M;
            float o = O[dn][r] / lrow[r];
            Zb[((size_t)b * S_LEN + q) * DM + h * DEPTH + dn * 16 + l16] = f2bf(o);
        }
    }
}

// ---------------------------------------------------------------------------
// Kernel 3: output projection. out = Z @ Wo + bo (fp32 out).
// grid = (B*S/64, DM/64), block = 128 (4 waves). A-tile via TDM when available.
// ---------------------------------------------------------------------------
__global__ __launch_bounds__(128) void out_proj_kernel(
    const __bf16* __restrict__ Zb, const float* __restrict__ Wo,
    const float* __restrict__ bo, float* __restrict__ out)
{
    __shared__ __attribute__((aligned(32))) __bf16 As[64][32];   // [row][k]
    __shared__ __attribute__((aligned(32))) __bf16 Wst[64][32];  // [col][k]

    const int tid  = threadIdx.x;
    const int wave = tid >> 5;
    const int lane = tid & 31;
    const int l16  = lane & 15;
    const int hi   = lane >> 4;

    const int rtile = blockIdx.x;      // 64-row tile over B*S
    const int ctile = blockIdx.y;      // 64-col tile over DM

    v8f acc[4];
#pragma unroll
    for (int m = 0; m < 4; ++m) acc[m] = (v8f){0,0,0,0,0,0,0,0};

    for (int k0 = 0; k0 < DM; k0 += 32) {
        __syncthreads();
        // As: Z tile [64 rows x 32 k], rows strided by DM in Zb
#if USE_TDM
        if (wave == 0) {
            tdm_load_2d_bf16(Zb + (size_t)(rtile * 64) * DM + k0, lds_off(&As[0][0]),
                             /*tile*/ 32, 64, /*tensor*/ 32, 64, /*stride*/ DM);
            __builtin_amdgcn_s_wait_tensorcnt(0);
        }
#else
#pragma unroll
        for (int i = 0; i < 2; ++i) {
            int t = tid + i * 128;               // 0..255
            int r = t >> 2, seg = t & 3;
            ((uint4*)&As[0][0])[t] =
                ((const uint4*)(Zb + (size_t)(rtile * 64 + r) * DM + k0))[seg];
        }
#endif
        // Wst[e][dl] = Wo[k0+dl][ctile*64+e], fp32 -> bf16
#pragma unroll
        for (int i = 0; i < 16; ++i) {
            int t = tid + i * 128;               // 0..2047
            int e = t >> 5, dl = t & 31;
            Wst[e][dl] = f2bf(Wo[(size_t)(k0 + dl) * DM + ctile * 64 + e]);
        }
        __syncthreads();

        v16bf bW = *(const v16bf*)&Wst[wave * 16 + l16][hi * 16];
#pragma unroll
        for (int m = 0; m < 4; ++m) {
            v16bf aA = load_a_frag(&As[m * 16 + l16][0], hi);
            acc[m] = wmma_bf16(aA, bW, acc[m]);
        }
    }

    const int colg = ctile * 64 + wave * 16 + l16;
    const float bb = bo[colg];
#pragma unroll
    for (int m = 0; m < 4; ++m) {
#pragma unroll
        for (int r = 0; r < 8; ++r) {
            int R = rtile * 64 + m * 16 + r + hi * 8;
            out[(size_t)R * DM + colg] = acc[m][r] + bb;
        }
    }
}

// ---------------------------------------------------------------------------
extern "C" void kernel_launch(void* const* d_in, const int* in_sizes, int n_in,
                              void* d_out, int out_size, void* d_ws, size_t ws_size,
                              hipStream_t stream) {
    const float* x    = (const float*)d_in[0];
    const float* mask = (const float*)d_in[1];
    const float* Wq   = (const float*)d_in[2];
    const float* bq   = (const float*)d_in[3];
    const float* Wk   = (const float*)d_in[4];
    const float* bk   = (const float*)d_in[5];
    const float* Wv   = (const float*)d_in[6];
    const float* bv   = (const float*)d_in[7];
    const float* Wo   = (const float*)d_in[8];
    const float* bo   = (const float*)d_in[9];
    float* out = (float*)d_out;

    // Workspace layout (bf16): Q | K | V^T | Z, 16 MB each (needs 64 MB total).
    const size_t NQ = (size_t)B_SZ * HEADS * S_LEN * DEPTH;   // 8M elems
    __bf16* Qb = (__bf16*)d_ws;
    __bf16* Kb = Qb + NQ;
    __bf16* Vt = Kb + NQ;
    __bf16* Zb = Vt + NQ;

    dim3 gA((B_SZ * S_LEN) / 64, HEADS, 3);
    qkv_proj_kernel<<<gA, 128, 0, stream>>>(x, Wq, bq, Wk, bk, Wv, bv, Qb, Kb, Vt);

    dim3 gB(B_SZ * HEADS, S_LEN / 64);
    attn_kernel<<<gB, 128, 0, stream>>>(Qb, Kb, Vt, mask, Zb);

    dim3 gC((B_SZ * S_LEN) / 64, DM / 64);
    out_proj_kernel<<<gC, 128, 0, stream>>>(Zb, Wo, bo, out);
}
